// MyModel_87522843560897
// MI455X (gfx1250) — matrix operand order repaired and checked by reading, not compiled, not run on hardware
//
#include <hip/hip_runtime.h>

#define VOCAB 100
#define EMBD  10
#define UNITS 32
#define DENSE 25
#define BATCH 4096
#define TSTEPS 512

#define NWAVES 4
#define HSTRIDE 34   // padded f32 row stride for the h transpose tile (even -> b64-aligned pairs)

#define LOG2E 1.44269504088896340736f

typedef __attribute__((ext_vector_type(16))) _Float16 v16h;
typedef __attribute__((ext_vector_type(8)))  float    v8f;

__device__ __forceinline__ v8f wmma16(v16h a, v16h b, v8f c) {
  // D = A(16x32 f16) * B(32x16 f16) + C(16x16 f32)
  return __builtin_amdgcn_wmma_f32_16x16x32_f16(false, a, false, b, (short)0, c,
                                                false, false);
}

// Build one 32x16 f16 B-tile in WMMA B layout from an LDS f32 matrix [32][stride].
// Lane l: N = ntile*16 + (l&15); koff = (l>>4)*8; VGPR v holds K = 2v+koff(+8 if v>=4), +1.
__device__ __forceinline__ v16h loadBtile(const float* S, int stride, int ntile, int lane) {
  v16h b;
  const int n    = ntile * 16 + (lane & 15);
  const int koff = (lane >> 4) * 8;
#pragma unroll
  for (int v = 0; v < 8; ++v) {
    b[2 * v]     = (_Float16)S[(2 * v + koff + ((v >= 4) ? 8 : 0)) * stride + n];
    b[2 * v + 1] = (_Float16)S[(2 * v + 1 + koff + ((v >= 4) ? 8 : 0)) * stride + n];
  }
  return b;
}

__global__ __launch_bounds__(NWAVES * 32, 1)
void gru_fused_kernel(const int* __restrict__ inp, const float* __restrict__ emb,
                      const float* __restrict__ Wk, const float* __restrict__ Wr,
                      const float* __restrict__ bias, const float* __restrict__ Wd,
                      const float* __restrict__ db, float* __restrict__ out)
{
  __shared__ float sWk[32 * 96];              // kernel, K zero-padded 10->32
  __shared__ float sWr[32 * 96];              // rec_kernel
  __shared__ float sWd[32 * 32];              // dense_w, N zero-padded 25->32
  __shared__ float sEmb[VOCAB * EMBD + 8];    // +8 pad: hi-half over-read stays in-array
  __shared__ float sH[NWAVES][16 * HSTRIDE];

  const int tid = threadIdx.x;
  for (int i = tid; i < 32 * 96; i += NWAVES * 32) {
    const int k = i / 96, n = i % 96;
    sWk[i] = (k < EMBD) ? Wk[k * 96 + n] : 0.0f;
    sWr[i] = Wr[i];
  }
  for (int i = tid; i < 32 * 32; i += NWAVES * 32) {
    const int k = i / 32, n = i % 32;
    sWd[i] = (n < DENSE) ? Wd[k * DENSE + n] : 0.0f;
  }
  for (int i = tid; i < VOCAB * EMBD + 8; i += NWAVES * 32)
    sEmb[i] = (i < VOCAB * EMBD) ? emb[i] : 0.0f;
  __syncthreads();

  const int lane = tid & 31;
  const int wave = tid >> 5;
  const int m    = lane & 15;   // row (A layout) / column (C layout) index
  const int hi   = lane >> 4;   // lane half
  const int b0   = blockIdx.x * (NWAVES * 16) + wave * 16;

  // Resident weight tiles (f16, WMMA B layout): 6 + 6 + 2 tiles
  v16h Bk[6], Br[6], Bd[2];
#pragma unroll
  for (int n = 0; n < 6; ++n) {
    Bk[n] = loadBtile(sWk, 96, n, lane);
    Br[n] = loadBtile(sWr, 96, n, lane);
  }
#pragma unroll
  for (int n = 0; n < 2; ++n) Bd[n] = loadBtile(sWd, 32, n, lane);

  // Per-lane scalar biases (column = N = lane%16 per tile).
  // z/r gates: (x-bias + rec-bias) pre-scaled by -log2e so sigmoid arg is one fma.
  float bzl[4], b0h[2], b1h[2], dbv[2];
#pragma unroll
  for (int n = 0; n < 4; ++n)
    bzl[n] = -(bias[n * 16 + m] + bias[96 + n * 16 + m]) * LOG2E;
#pragma unroll
  for (int n = 0; n < 2; ++n) {
    b0h[n] = bias[(n + 4) * 16 + m];
    b1h[n] = bias[96 + (n + 4) * 16 + m];
  }
  dbv[0] = db[m];
  dbv[1] = (16 + m < DENSE) ? db[16 + m] : 0.0f;

  const v8f Z0 = {0.0f, 0.0f, 0.0f, 0.0f, 0.0f, 0.0f, 0.0f, 0.0f};

  // h state: C-layout f32 (2 N-tiles x 8 rows) and A-layout f16 (16 K per lane)
  float hC[2][8];
#pragma unroll
  for (int n = 0; n < 2; ++n)
#pragma unroll
    for (int r = 0; r < 8; ++r) hC[n][r] = 0.0f;

  v16h Ah, Ax;
#pragma unroll
  for (int j = 0; j < 16; ++j) { Ah[j] = (_Float16)0.0f; Ax[j] = (_Float16)0.0f; }
  // Ax[8..15] (K=16..31 zero-pad) stays invariant for the whole loop.

  const int* tokp = inp + (size_t)(b0 + m) * TSTEPS;
  float* sHw = sH[wave];

  // Two per-lane output pointers; row offsets become constant immediate offsets.
  float* o0 = out + (size_t)(b0 + 8 * hi) * (TSTEPS * DENSE) + m;
  float* o1 = o0 + 16;

  for (int t = 0; t < TSTEPS; ++t) {
    // ---- x_t = emb[tok] in A layout, branchless (lanes0-15: K0..7, lanes16-31: K8..9) ----
    const int tok = tokp[t];
    __builtin_prefetch(tokp + t + 64, 0, 3);   // global_prefetch of token stream
    const float2* e2 = (const float2*)(&sEmb[tok * EMBD + 8 * hi]);
#pragma unroll
    for (int p = 0; p < 4; ++p) {
      const float2 f = e2[p];
      const _Float16 a0 = (_Float16)f.x;
      const _Float16 a1 = (_Float16)f.y;
      const bool keep = (hi == 0) || (p == 0);   // hi half only has emb cols 8,9
      Ax[2 * p]     = keep ? a0 : (_Float16)0.0f;
      Ax[2 * p + 1] = keep ? a1 : (_Float16)0.0f;
    }

    // ---- gates; z/r pre-activations accumulate x-part via chained C operand ----
#pragma unroll
    for (int n = 0; n < 2; ++n) {   // two 16-column tiles cover U=32
      v8f zp = wmma16(Ah, Br[n],     Z0);   // h @ Wr_z
      v8f rp = wmma16(Ah, Br[n + 2], Z0);   // h @ Wr_r
      v8f xh = wmma16(Ax, Bk[n + 4], Z0);   // x @ Wk_h
      v8f rh = wmma16(Ah, Br[n + 4], Z0);   // h @ Wr_h
      zp = wmma16(Ax, Bk[n],     zp);       // += x @ Wk_z
      rp = wmma16(Ax, Bk[n + 2], rp);       // += x @ Wk_r
#pragma unroll
      for (int r = 0; r < 8; ++r) {
        // sigmoid(x+b) = rcp(1 + exp2(fma(x, -log2e, -b*log2e)))
        const float z  = __builtin_amdgcn_rcpf(
            1.0f + __builtin_amdgcn_exp2f(fmaf(zp[r], -LOG2E, bzl[n])));
        const float rg = __builtin_amdgcn_rcpf(
            1.0f + __builtin_amdgcn_exp2f(fmaf(rp[r], -LOG2E, bzl[n + 2])));
        // tanh(x) = 1 - 2*rcp(1 + exp2(2*log2e*x))
        const float x  = fmaf(rg, rh[r] + b1h[n], xh[r] + b0h[n]);
        const float hh = fmaf(-2.0f,
            __builtin_amdgcn_rcpf(1.0f + __builtin_amdgcn_exp2f(x * (2.0f * LOG2E))),
            1.0f);
        hC[n][r] = fmaf(z, hC[n][r] - hh, hh);
      }
    }

    // ---- C-layout -> A-layout transpose of h through per-wave LDS tile ----
#pragma unroll
    for (int n = 0; n < 2; ++n)
#pragma unroll
      for (int r = 0; r < 8; ++r)
        sHw[(r + 8 * hi) * HSTRIDE + n * 16 + m] = hC[n][r];

    asm volatile("s_wait_dscnt 0" ::: "memory");  // same-wave LDS RAW (in-order DS pipe)

    {
      const float2* hrow = (const float2*)(&sHw[m * HSTRIDE + 8 * hi]);
#pragma unroll
      for (int v = 0; v < 8; ++v) {
        const float2 f = hrow[(v < 4) ? v : (v + 4)];  // K = 2v+8hi / 16+2(v-4)+8hi
        Ah[2 * v]     = (_Float16)f.x;
        Ah[2 * v + 1] = (_Float16)f.y;
      }
    }

    // ---- fused dense + ReLU + store: y = relu(h @ Wd + db) ----
    {
      v8f y0 = wmma16(Ah, Bd[0], Z0);
#pragma unroll
      for (int r = 0; r < 8; ++r)
        o0[(size_t)r * (TSTEPS * DENSE)] = fmaxf(y0[r] + dbv[0], 0.0f);
      v8f y1 = wmma16(Ah, Bd[1], Z0);
      if (m < 9) {
#pragma unroll
        for (int r = 0; r < 8; ++r)
          o1[(size_t)r * (TSTEPS * DENSE)] = fmaxf(y1[r] + dbv[1], 0.0f);
      }
    }
    o0 += DENSE;
    o1 += DENSE;
  }
}

extern "C" void kernel_launch(void* const* d_in, const int* in_sizes, int n_in,
                              void* d_out, int out_size, void* d_ws, size_t ws_size,
                              hipStream_t stream) {
  (void)in_sizes; (void)n_in; (void)out_size; (void)d_ws; (void)ws_size;
  const int*   inp  = (const int*)d_in[0];
  const float* emb  = (const float*)d_in[1];
  const float* Wk   = (const float*)d_in[2];
  const float* Wr   = (const float*)d_in[3];
  const float* bias = (const float*)d_in[4];
  const float* Wd   = (const float*)d_in[5];
  const float* db   = (const float*)d_in[6];
  float* out = (float*)d_out;

  dim3 grid(BATCH / (NWAVES * 16));   // 64 blocks -> 256 waves, one 16-row tile each
  dim3 block(NWAVES * 32);
  hipLaunchKernelGGL(gru_fused_kernel, grid, block, 0, stream,
                     inp, emb, Wk, Wr, bias, Wd, db, out);
}
